// vanillaMLP_48988396978336
// MI455X (gfx1250) — matrix operand order repaired and checked by reading, not compiled, not run on hardware
//
#include <hip/hip_runtime.h>
#include <hip/hip_bf16.h>
#include <math.h>

typedef __attribute__((ext_vector_type(16))) _Float16 v16h;
typedef __attribute__((ext_vector_type(8)))  float    v8f;

#define T_SIZE   (1u << 19)
#define T_MASK   (T_SIZE - 1u)
#define N_ACTIVE_LEVELS 4          // MASK kills features >= 8 -> levels >= 4 never matter
#define DENSITY_BIAS (-1.0f)
#define CLAMP_MAX 10.0f

// floor(16 * b^l), b = exp(ln(2048/16)/15): levels 0..3
__device__ __constant__ float kRes[N_ACTIVE_LEVELS] = {16.f, 22.f, 30.f, 42.f};

// A-fragment K index for 16-bit A 16x32 layout (ISA 7.12.2):
// lanes 0-15: VGPR0..3 -> K0..7, VGPR4..7 -> K16..23 ; lanes 16-31: K8..15, K24..31
__device__ __forceinline__ int a_kmap(int e, int hl) {
    int v = e >> 1, j = e & 1;
    return (v < 4) ? ((hl ? 8 : 0) + 2 * v + j)
                   : ((hl ? 24 : 16) + 2 * (v - 4) + j);
}

__global__ __launch_bounds__(256)
void nerf_fused_fwdbwd(const float* __restrict__ pts,
                       const float* __restrict__ table,   // [16][2^19][2] f32
                       const float* __restrict__ W1,      // [64][32] f32
                       const float* __restrict__ W2,      // [16][64] f32
                       float* __restrict__ out, int n)
{
    __shared__ _Float16 h_lds [8][16][64];  // relu(h), per-wave slice, 16 KB
    __shared__ float    f0_lds[8][16];      // fea[:,0] per point
    __shared__ float    df_lds[8][16][8];   // masked dfeat per point

    const int lane = threadIdx.x & 31;
    const int wave = threadIdx.x >> 5;
    const int hl   = lane >> 4;      // which 16-lane half
    const int lp   = lane & 15;      // point index within the wave's 16-point tile
    const int base = (blockIdx.x * 8 + wave) * 16;

    int p = base + lp; if (p >= n) p = n - 1;   // clamp loads; stores guarded

    // normalized coords in [0,1]
    const float x0 = (pts[(size_t)p * 3 + 0] + 1.f) * 0.5f;
    const float x1 = (pts[(size_t)p * 3 + 1] + 1.f) * 0.5f;
    const float x2 = (pts[(size_t)p * 3 + 2] + 1.f) * 0.5f;

    // ---------------- forward hash encoding: this lane does 2 levels ----------------
    const int lb = hl * 2;                    // lanes 0-15: levels 0,1 ; lanes 16-31: 2,3
    float feat[4];
    #pragma unroll
    for (int li = 0; li < 2; ++li) {
        const int   l   = lb + li;
        const float res = kRes[l];
        const float px = x0 * res, py = x1 * res, pz = x2 * res;
        const float fx = floorf(px), fy = floorf(py), fz = floorf(pz);
        const float rx = px - fx,  ry = py - fy,  rz = pz - fz;
        const unsigned ix = (unsigned)(int)fx, iy = (unsigned)(int)fy, iz = (unsigned)(int)fz;
        const float* tb = table + (size_t)l * T_SIZE * 2;
        float f0 = 0.f, f1 = 0.f;
        #pragma unroll
        for (int c = 0; c < 8; ++c) {
            const unsigned ox = (c >> 2) & 1u, oy = (c >> 1) & 1u, oz = c & 1u;
            const unsigned h  = (ix + ox) ^ ((iy + oy) * 2654435761u) ^ ((iz + oz) * 805459861u);
            const unsigned idx = h & T_MASK;
            const float wx = ox ? rx : 1.f - rx;
            const float wy = oy ? ry : 1.f - ry;
            const float wz = oz ? rz : 1.f - rz;
            const float w  = wx * wy * wz;
            f0 = fmaf(w, tb[(size_t)idx * 2 + 0], f0);
            f1 = fmaf(w, tb[(size_t)idx * 2 + 1], f1);
        }
        feat[li * 2 + 0] = f0; feat[li * 2 + 1] = f1;
    }
    float of[4];                               // partner half's 2 levels
    #pragma unroll
    for (int i = 0; i < 4; ++i) of[i] = __shfl_xor(feat[i], 16, 32);

    // ---------------- GEMM1: h = relu(feat @ W1^T), 4 WMMAs ----------------
    // A: lanes 0-15 hold K0..7 = features; lanes 16-31 hold K8..15 = masked zeros.
    v16h a1;
    #pragma unroll
    for (int e = 0; e < 16; ++e) a1[e] = (_Float16)0.f;
    if (hl == 0) {
        #pragma unroll
        for (int i = 0; i < 4; ++i) { a1[i] = (_Float16)feat[i]; a1[4 + i] = (_Float16)of[i]; }
    }
    #pragma unroll
    for (int t = 0; t < 4; ++t) {              // neuron tiles n0 = 16*t
        v16h b;
        #pragma unroll
        for (int e = 0; e < 16; ++e) b[e] = (_Float16)0.f;
        if (hl == 0) {                         // lane half 0 holds K0..7, K16..23; only K<8 nonzero
            const float* wrow = W1 + (size_t)(t * 16 + lp) * 32;
            #pragma unroll
            for (int e = 0; e < 8; ++e) b[e] = (_Float16)wrow[e];
        }
        v8f c;
        #pragma unroll
        for (int r = 0; r < 8; ++r) c[r] = 0.f;
        c = __builtin_amdgcn_wmma_f32_16x16x32_f16(false, a1, false, b, (short)0, c, false, false);
        #pragma unroll
        for (int r = 0; r < 8; ++r) {
            const float hv = fmaxf(c[r], 0.f);                 // relu
            h_lds[wave][r + 8 * hl][t * 16 + lp] = (_Float16)hv; // D layout: M=r+8*hl, N=lp
        }
    }
    __syncthreads();

    // ---------------- GEMM2: fea = h @ W2^T, 2 chained WMMAs (K=64) ----------------
    v8f fea;
    #pragma unroll
    for (int r = 0; r < 8; ++r) fea[r] = 0.f;
    #pragma unroll
    for (int kt = 0; kt < 2; ++kt) {
        v16h a2, b2;
        #pragma unroll
        for (int e = 0; e < 16; ++e) {
            const int K = a_kmap(e, hl) + kt * 32;
            a2[e] = h_lds[wave][lp][K];
        }
        #pragma unroll
        for (int e = 0; e < 16; ++e) {
            const int K = a_kmap(e, hl) + kt * 32;             // B: same K order as A
            b2[e] = (_Float16)W2[(size_t)lp * 64 + K];         // B[k][n] = W2[n][k], n = lp
        }
        fea = __builtin_amdgcn_wmma_f32_16x16x32_f16(false, a2, false, b2, (short)0, fea, false, false);
    }
    // write fea [n,16]; stash fea0 for sigma/backward
    float* out_fea = out + (size_t)n;
    #pragma unroll
    for (int r = 0; r < 8; ++r) {
        const int M = r + 8 * hl, pp = base + M;
        if (pp < n) out_fea[(size_t)pp * 16 + lp] = fea[r];
    }
    if (lp == 0) {
        #pragma unroll
        for (int r = 0; r < 8; ++r) f0_lds[wave][r + 8 * hl] = fea[r];
    }
    __syncthreads();

    const float fea0 = f0_lds[wave][lp];
    const float sgrad = expf(fminf(fea0 + DENSITY_BIAS, CLAMP_MAX));  // trunc_exp bwd
    if (hl == 0 && (base + lp) < n) out[base + lp] = expf(fea0 + DENSITY_BIAS); // sigma

    // ---------------- GEMM3: dfeat = dh @ W1, 2 chained WMMAs ----------------
    // dh[p][k] = sgrad_p * W2[0][k] * (h[p][k] > 0)
    v8f df;
    #pragma unroll
    for (int r = 0; r < 8; ++r) df[r] = 0.f;
    #pragma unroll
    for (int kt = 0; kt < 2; ++kt) {
        v16h a3, b3;
        #pragma unroll
        for (int e = 0; e < 16; ++e) {
            const int K = a_kmap(e, hl) + kt * 32;
            const float hv = (float)h_lds[wave][lp][K];
            a3[e] = (_Float16)((hv > 0.f) ? sgrad * W2[K] : 0.f);
        }
        #pragma unroll
        for (int e = 0; e < 16; ++e) {
            const int K = a_kmap(e, hl) + kt * 32;
            b3[e] = (_Float16)W1[(size_t)K * 32 + lp];         // B[k(neuron)][n(feat)] = W1[k][n]
        }
        df = __builtin_amdgcn_wmma_f32_16x16x32_f16(false, a3, false, b3, (short)0, df, false, false);
    }
    if (lp < 8) {                                              // only masked 8 cols matter
        #pragma unroll
        for (int r = 0; r < 8; ++r) df_lds[wave][r + 8 * hl][lp] = df[r];
    }
    __syncthreads();

    // ---------------- backward encoding: re-gather (L2-resident) ----------------
    float gx = 0.f, gy = 0.f, gz = 0.f;
    #pragma unroll
    for (int li = 0; li < 2; ++li) {
        const int   l   = lb + li;
        const float res = kRes[l];
        const float d0 = df_lds[wave][lp][2 * l + 0];
        const float d1 = df_lds[wave][lp][2 * l + 1];
        const float px = x0 * res, py = x1 * res, pz = x2 * res;
        const float fx = floorf(px), fy = floorf(py), fz = floorf(pz);
        const float rx = px - fx,  ry = py - fy,  rz = pz - fz;
        const unsigned ix = (unsigned)(int)fx, iy = (unsigned)(int)fy, iz = (unsigned)(int)fz;
        const float* tb = table + (size_t)l * T_SIZE * 2;
        float lgx = 0.f, lgy = 0.f, lgz = 0.f;
        #pragma unroll
        for (int c = 0; c < 8; ++c) {
            const unsigned ox = (c >> 2) & 1u, oy = (c >> 1) & 1u, oz = c & 1u;
            const unsigned h  = (ix + ox) ^ ((iy + oy) * 2654435761u) ^ ((iz + oz) * 805459861u);
            const unsigned idx = h & T_MASK;
            const float wx = ox ? rx : 1.f - rx;
            const float wy = oy ? ry : 1.f - ry;
            const float wz = oz ? rz : 1.f - rz;
            const float g  = d0 * tb[(size_t)idx * 2 + 0] + d1 * tb[(size_t)idx * 2 + 1];
            const float sx = ox ? 1.f : -1.f, sy = oy ? 1.f : -1.f, sz = oz ? 1.f : -1.f;
            lgx = fmaf(g, sx * wy * wz, lgx);
            lgy = fmaf(g, sy * wx * wz, lgy);
            lgz = fmaf(g, sz * wx * wy, lgz);
        }
        gx = fmaf(lgx, res, gx); gy = fmaf(lgy, res, gy); gz = fmaf(lgz, res, gz);
    }
    // sum the two-level halves across the lane pair
    gx += __shfl_xor(gx, 16, 32);
    gy += __shfl_xor(gy, 16, 32);
    gz += __shfl_xor(gz, 16, 32);

    if (hl == 0 && (base + lp) < n) {
        const size_t pp = (size_t)(base + lp);
        const float inv = 1.f / sqrtf(gx * gx + gy * gy + gz * gz);
        float* out_nrm  = out + 17 * (size_t)n;
        float* out_grad = out + 20 * (size_t)n;
        out_nrm [pp * 3 + 0] = gx * inv;
        out_nrm [pp * 3 + 1] = gy * inv;
        out_nrm [pp * 3 + 2] = gz * inv;
        out_grad[pp * 3 + 0] = gx;
        out_grad[pp * 3 + 1] = gy;
        out_grad[pp * 3 + 2] = gz;
    }
}

extern "C" void kernel_launch(void* const* d_in, const int* in_sizes, int n_in,
                              void* d_out, int out_size, void* d_ws, size_t ws_size,
                              hipStream_t stream) {
    const float* pts   = (const float*)d_in[0];
    const float* table = (const float*)d_in[1];
    const float* W1    = (const float*)d_in[2];
    const float* W2    = (const float*)d_in[3];
    float* out = (float*)d_out;
    const int n = in_sizes[0] / 3;                 // pts is [n,3]
    const int blocks = (n + 127) / 128;            // 8 waves * 16 points per block
    hipLaunchKernelGGL(nerf_fused_fwdbwd, dim3(blocks), dim3(256), 0, stream,
                       pts, table, W1, W2, out, n);
}
